// HiddenMarkovModel_77781857730638
// MI455X (gfx1250) — matrix operand order repaired and checked by reading, not compiled, not run on hardware
//
#include <hip/hip_runtime.h>
#include <hip/hip_bf16.h>

typedef __attribute__((ext_vector_type(16))) _Float16 v16h;
typedef __attribute__((ext_vector_type(8)))  float    v8f;

#define NUM_S   2048
#define NUM_T   40
#define NUM_P   12341
#define P_PAD   12352           // 16*772
#define P_TILES 772
#define S_TILES 128
#define ROW_BYTES 320           // 160 f16 per row (T*4 = K dim)
#define GRID_Y  16              // 16 blocks * 8 waves = 128 wave-slots per s-tile

// ---- workspace layout (bytes) ----
#define WS_LOGSR   0                                  // 16 f32
#define WS_LOGPRI  64                                 // 4  f32
#define WS_LOGTR   80                                 // 16 f32
#define WS_LPP     8448                               // P_PAD f32
#define WS_ONEHOT  57856                              // P_PAD * 320  (f16 one-hot, N-major "B^T")
#define WS_BHI     (WS_ONEHOT + (size_t)P_PAD * ROW_BYTES)   // 2048*320 f16 hi-split
#define WS_BLO     (WS_BHI   + (size_t)NUM_S * ROW_BYTES)    // 2048*320 f16 lo-split
#define WS_PARTIAL (WS_BLO   + (size_t)NUM_S * ROW_BYTES)    // 128 stiles * 128 slots * 16 f32
#define WS_DENOM   (WS_PARTIAL + (size_t)S_TILES * 128 * 16 * 4) // 2048 f32

static __device__ inline unsigned short h2u(_Float16 h) {
  union { _Float16 h; unsigned short u; } c; c.h = h; return c.u;
}

// ---------------- parameter transforms ----------------
__global__ void hmm_prep(const float* __restrict__ wss,
                         const float* __restrict__ wsp,
                         const float* __restrict__ wtr,
                         char* __restrict__ ws) {
  if (threadIdx.x != 0) return;
  float* logsr  = (float*)(ws + WS_LOGSR);
  float* logpri = (float*)(ws + WS_LOGPRI);
  float* logtr  = (float*)(ws + WS_LOGTR);

  // softmax(w_strategy_success + triu(-inf,k=1)) row-wise: row i over j<=i
  float pss[4][4];
  for (int i = 0; i < 4; ++i) {
    float e[4] = {0.f, 0.f, 0.f, 0.f};
    float mx = -1e30f;
    for (int j = 0; j <= i; ++j) mx = fmaxf(mx, wss[i*4+j]);
    float den = 0.f;
    for (int j = 0; j <= i; ++j) { e[j] = expf(wss[i*4+j] - mx); den += e[j]; }
    for (int j = 0; j < 4; ++j) pss[i][j] = (j <= i) ? e[j] / den : 0.f;
  }
  const float psr[4][4] = {{3.f/9.f, 4.f/9.f, 1.f/9.f, 1.f/9.f},
                           {0.f,     4.f/6.f, 1.f/6.f, 1.f/6.f},
                           {0.f,     0.f,     0.5f,    0.5f   },
                           {0.f,     0.f,     0.f,     1.f    }};
  for (int i = 0; i < 4; ++i)
    for (int j = 0; j < 4; ++j) {
      float v = 0.f;
      for (int k = 0; k < 4; ++k) v += pss[i][k] * psr[k][j];
      logsr[i*4+j] = logf(v);
    }
  // log softmax of priors
  {
    float mx = -1e30f;
    for (int i = 0; i < 4; ++i) mx = fmaxf(mx, wsp[i]);
    float den = 0.f;
    for (int i = 0; i < 4; ++i) den += expf(wsp[i] - mx);
    float ld = logf(den);
    for (int i = 0; i < 4; ++i) logpri[i] = wsp[i] - mx - ld;
  }
  // log softmax(w_transitions + tril(-inf,k=-1)) row-wise: row i over j>=i
  for (int i = 0; i < 4; ++i) {
    float mx = -1e30f;
    for (int j = i; j < 4; ++j) mx = fmaxf(mx, wtr[i*4+j]);
    float den = 0.f;
    for (int j = i; j < 4; ++j) den += expf(wtr[i*4+j] - mx);
    float ld = logf(den);
    for (int j = 0; j < 4; ++j)
      logtr[i*4+j] = (j >= i) ? (wtr[i*4+j] - mx - ld) : -1e30f;
  }
}

// ---------------- per-path: log p(path) + one-hot f16 row (B^T, N-major) ----------------
__global__ void hmm_paths(const int* __restrict__ paths, char* __restrict__ ws) {
  int p = blockIdx.x * blockDim.x + threadIdx.x;
  if (p >= P_PAD) return;
  float* lpp = (float*)(ws + WS_LPP);
  uint2* oh  = (uint2*)(ws + WS_ONEHOT + (size_t)p * ROW_BYTES);
  if (p >= NUM_P) {                       // padding: zero one-hot, -inf-ish prior
    lpp[p] = -1e30f;
    for (int t = 0; t < NUM_T; ++t) oh[t] = make_uint2(0u, 0u);
    return;
  }
  const float* logpri = (const float*)(ws + WS_LOGPRI);
  const float* logtr  = (const float*)(ws + WS_LOGTR);
  const int* row = paths + (size_t)p * NUM_T;
  int prev = row[0];
  float lp = logpri[prev];
  {
    unsigned h0 = (prev==0 ? 0x3C00u : 0u) | ((prev==1 ? 0x3C00u : 0u) << 16);
    unsigned h1 = (prev==2 ? 0x3C00u : 0u) | ((prev==3 ? 0x3C00u : 0u) << 16);
    oh[0] = make_uint2(h0, h1);
  }
  for (int t = 1; t < NUM_T; ++t) {
    int st = row[t];
    lp += logtr[prev*4 + st];
    unsigned h0 = (st==0 ? 0x3C00u : 0u) | ((st==1 ? 0x3C00u : 0u) << 16);
    unsigned h1 = (st==2 ? 0x3C00u : 0u) | ((st==3 ? 0x3C00u : 0u) << 16);
    oh[t] = make_uint2(h0, h1);
    prev = st;
  }
  lpp[p] = lp;
}

// ---------------- per-subject: split-f16 log-response matrix ----------------
__global__ void hmm_subj(const int* __restrict__ resp, char* __restrict__ ws) {
  int idx = blockIdx.x * blockDim.x + threadIdx.x;
  if (idx >= NUM_S * NUM_T) return;
  int s = idx / NUM_T, t = idx % NUM_T;
  const float* logsr = (const float*)(ws + WS_LOGSR);
  int r = resp[idx];
  unsigned hhi[2] = {0u, 0u}, hlo[2] = {0u, 0u};
  for (int k = 0; k < 4; ++k) {
    float f = logsr[k*4 + r];
    _Float16 hi = (_Float16)f;
    _Float16 lo = (_Float16)(f - (float)hi);
    hhi[k>>1] |= (unsigned)h2u(hi) << ((k & 1) * 16);
    hlo[k>>1] |= (unsigned)h2u(lo) << ((k & 1) * 16);
  }
  uint2* bhi = (uint2*)(ws + WS_BHI + (size_t)s * ROW_BYTES);
  uint2* blo = (uint2*)(ws + WS_BLO + (size_t)s * ROW_BYTES);
  bhi[t] = make_uint2(hhi[0], hhi[1]);
  blo[t] = make_uint2(hlo[0], hlo[1]);
}

// A-operand fragment (16x32 f16, ISA interleaved layout): two b128 at +0 and +32
static __device__ inline v16h load_frag_a(const char* base) {
  union { uint4 u[2]; v16h v; } r;
  r.u[0] = *(const uint4*)(base);
  r.u[1] = *(const uint4*)(base + 32);
  return r.v;
}
// B-operand fragment (32x16 f16, N-major source): 32 contiguous bytes
static __device__ inline v16h load_frag_b(const char* base) {
  union { uint4 u[2]; v16h v; } r;
  r.u[0] = *(const uint4*)(base);
  r.u[1] = *(const uint4*)(base + 16);
  return r.v;
}

// ---------------- WMMA GEMM + exp + partial sums ----------------
__global__ void __launch_bounds__(256) hmm_gemm(char* __restrict__ ws,
                                                float* __restrict__ out) {
  const int lane = threadIdx.x & 31;
  const int wv   = threadIdx.x >> 5;     // wave in block, 0..7
  const int h    = lane >> 4;            // half-wave
  const int m    = lane & 15;
  const int s_base = blockIdx.x * 16;

  // A fragments: 16 subjects x 32 K per chunk; lane row = s_base+m.
  // ISA 16-bit A layout: h=0 -> K=[0..7]+[16..23]; h=1 -> K=[8..15]+[24..31]
  const char* arow_hi = ws + WS_BHI + (size_t)(s_base + m) * ROW_BYTES;
  const char* arow_lo = ws + WS_BLO + (size_t)(s_base + m) * ROW_BYTES;
  v16h ahi[5], alo[5];
#pragma unroll
  for (int c = 0; c < 5; ++c) {
    ahi[c] = load_frag_a(arow_hi + c*64 + h*16);
    alo[c] = load_frag_a(arow_lo + c*64 + h*16);
  }

  const float* lpp = (const float*)(ws + WS_LPP);
  float acc[8];
#pragma unroll
  for (int r = 0; r < 8; ++r) acc[r] = 0.f;

  const int wslot = blockIdx.y * 8 + wv;           // 0..127
  for (int pt = wslot; pt < P_TILES; pt += GRID_Y * 8) {
    const int p = pt * 16 + m;                     // this lane's path column
    const char* brow = ws + WS_ONEHOT + (size_t)p * ROW_BYTES;
    v16h b[5];
#pragma unroll
    for (int k = 0; k < 5; ++k) b[k] = load_frag_a(brow + k*64 + h*16);

    v8f c = {};
#pragma unroll
    for (int k = 0; k < 5; ++k)
      c = __builtin_amdgcn_wmma_f32_16x16x32_f16(false, ahi[k], false, b[k],
                                                 (short)0, c, false, false);
#pragma unroll
    for (int k = 0; k < 5; ++k)
      c = __builtin_amdgcn_wmma_f32_16x16x32_f16(false, alo[k], false, b[k],
                                                 (short)0, c, false, false);

    const bool  valid = (p < NUM_P);
    const float lp    = lpp[p];                    // -1e30 on padded rows
    float num[8];
#pragma unroll
    for (int r = 0; r < 8; ++r) {                  // D: subject = s_base+r+8h, path col = m
      float v = __expf(c[r] + lp);                 // v_mul + v_exp_f32
      v = valid ? v : 0.f;
      num[r] = v;
      acc[r] += v;
    }
    if (valid) {                                   // single exec toggle per tile
      float* col = out + (size_t)(s_base + 8*h) * NUM_P + p;
#pragma unroll
      for (int r = 0; r < 8; ++r)
        col[(size_t)r * NUM_P] = num[r];
    }
  }

  // reduce over the 16 path columns (within each 16-lane half)
#pragma unroll
  for (int r = 0; r < 8; ++r) {
    float v = acc[r];
    v += __shfl_xor(v, 1, 32);
    v += __shfl_xor(v, 2, 32);
    v += __shfl_xor(v, 4, 32);
    v += __shfl_xor(v, 8, 32);
    acc[r] = v;
  }
  if (m == 0) {                                    // lanes 0 and 16
    float* dst = (float*)(ws + WS_PARTIAL) +
                 ((size_t)blockIdx.x * 128 + wslot) * 16;
#pragma unroll
    for (int r = 0; r < 8; ++r) dst[r + 8*h] = acc[r];
  }
}

// ---------------- deterministic denominator reduction ----------------
__global__ void hmm_reduce(char* __restrict__ ws) {
  int s = blockIdx.x * blockDim.x + threadIdx.x;
  if (s >= NUM_S) return;
  const float* partial = (const float*)(ws + WS_PARTIAL);
  int stile = s >> 4, sin = s & 15;
  float a = 0.f;
  for (int w = 0; w < 128; ++w)
    a += partial[((size_t)stile * 128 + w) * 16 + sin];
  ((float*)(ws + WS_DENOM))[s] = a;
}

// ---------------- row normalization ----------------
__global__ void hmm_norm(float* __restrict__ out, const char* __restrict__ ws) {
  const float inv = 1.0f / ((const float*)(ws + WS_DENOM))[blockIdx.x];
  float* row = out + (size_t)blockIdx.x * NUM_P;
  for (int p = threadIdx.x; p < NUM_P; p += blockDim.x) row[p] *= inv;
}

extern "C" void kernel_launch(void* const* d_in, const int* in_sizes, int n_in,
                              void* d_out, int out_size, void* d_ws, size_t ws_size,
                              hipStream_t stream) {
  const int*   resp  = (const int*)d_in[0];   // [2048,40]
  const int*   paths = (const int*)d_in[1];   // [12341,40]
  const float* wss   = (const float*)d_in[2]; // [4,4]
  const float* wsp   = (const float*)d_in[3]; // [4]
  const float* wtr   = (const float*)d_in[4]; // [4,4]
  char*  ws  = (char*)d_ws;
  float* out = (float*)d_out;

  hmm_prep <<<1, 64, 0, stream>>>(wss, wsp, wtr, ws);
  hmm_paths<<<(P_PAD + 127) / 128, 128, 0, stream>>>(paths, ws);
  hmm_subj <<<(NUM_S * NUM_T + 255) / 256, 256, 0, stream>>>(resp, ws);
  dim3 g(S_TILES, GRID_Y);
  hmm_gemm <<<g, 256, 0, stream>>>(ws, out);
  hmm_reduce<<<(NUM_S + 255) / 256, 256, 0, stream>>>(ws);
  hmm_norm <<<NUM_S, 256, 0, stream>>>(out, ws);
}
